// sLSTMBlock_67061619359782
// MI455X (gfx1250) — matrix operand order repaired and checked by reading, not compiled, not run on hardware
//
#include <hip/hip_runtime.h>
#include <hip/hip_bf16.h>
#include <math.h>

// ---------------------------------------------------------------------------
// Problem constants
// ---------------------------------------------------------------------------
#define BB    8192
#define DD    1024
#define HH    1024
#define NHH   8
#define DB    128          // per-head block size
#define PROJ  1365
#define PROJP 1376         // padded to 43 * 32

// ---------------------------------------------------------------------------
// WMMA types / helpers (gfx1250, wave32, v_wmma_f32_16x16x32_bf16)
// ---------------------------------------------------------------------------
typedef __attribute__((ext_vector_type(16))) __bf16         v16bf;
typedef __attribute__((ext_vector_type(8)))  float          v8f;
typedef __attribute__((ext_vector_type(8)))  unsigned short us8;
typedef __attribute__((ext_vector_type(16))) unsigned short us16;

union BFCast { us16 u; v16bf b; };

__device__ __forceinline__ unsigned short f2bf(float f) {
  unsigned int u = __float_as_uint(f);
  u += 0x7fffu + ((u >> 16) & 1u);            // round-to-nearest-even
  return (unsigned short)(u >> 16);
}

__device__ __forceinline__ v16bf make_frag(us8 lo, us8 hi) {
  BFCast c;
#pragma unroll
  for (int e = 0; e < 8; ++e) { c.u[e] = lo[e]; c.u[8 + e] = hi[e]; }
  return c.b;
}

// A fragment: src row-major [rows][ld] bf16, 16x32 tile at (row, kbase).
// Per ISA layout: lane half 'hi' holds K in [hi*8, hi*8+8) and [16+hi*8, ...).
__device__ __forceinline__ v16bf load_a(const unsigned short* __restrict__ p,
                                        int ld, int row, int kbase, int hi) {
  const unsigned short* q = p + (long)row * ld + kbase;
  return make_frag(*(const us8*)(q + hi * 8), *(const us8*)(q + 16 + hi * 8));
}

// B fragment: src packed [N][ldK] bf16 (K contiguous). Lane column n,
// half 'hi' holds K in [kbase + hi*8, +8) and [kbase + 16 + hi*8, +8).
__device__ __forceinline__ v16bf load_b(const unsigned short* __restrict__ p,
                                        int ld, int n, int kbase, int hi) {
  const unsigned short* q = p + (long)n * ld + kbase;
  return make_frag(*(const us8*)(q + hi * 8), *(const us8*)(q + 16 + hi * 8));
}

#define WMMA_BF16(a, b, c) \
  __builtin_amdgcn_wmma_f32_16x16x32_bf16(false, (a), false, (b), (short)0, (c), false, false)

__device__ __forceinline__ float sigmoidf_(float v) { return 1.f / (1.f + expf(-v)); }

// ---------------------------------------------------------------------------
// Kernel 1: LayerNorm + causal conv1d(4) + SiLU.  Emits bf16(x) and bf16(x_conv).
// One block per row, 256 threads, D=1024.
// ---------------------------------------------------------------------------
__global__ __launch_bounds__(256) void prep_x_kernel(
    const float* __restrict__ x, const float* __restrict__ ln_g,
    const float* __restrict__ ln_b, const float* __restrict__ conv_w,
    const float* __restrict__ conv_b, unsigned short* __restrict__ xb,
    unsigned short* __restrict__ xcb) {
  __shared__ float xn[DD];
  __shared__ float red[256];
  const int row = blockIdx.x;
  const int t = threadIdx.x;
  const float* xr = x + (long)row * DD;

  float v[4];
  float s = 0.f;
#pragma unroll
  for (int i = 0; i < 4; ++i) {
    v[i] = xr[t + 256 * i];
    s += v[i];
    xb[(long)row * DD + t + 256 * i] = f2bf(v[i]);   // raw x in bf16 (z/o gates)
  }
  red[t] = s;
  __syncthreads();
  for (int off = 128; off > 0; off >>= 1) {
    if (t < off) red[t] += red[t + off];
    __syncthreads();
  }
  const float mu = red[0] * (1.f / (float)DD);
  __syncthreads();
  float sq = 0.f;
#pragma unroll
  for (int i = 0; i < 4; ++i) { float d = v[i] - mu; sq += d * d; }
  red[t] = sq;
  __syncthreads();
  for (int off = 128; off > 0; off >>= 1) {
    if (t < off) red[t] += red[t + off];
    __syncthreads();
  }
  const float rstd = rsqrtf(red[0] * (1.f / (float)DD) + 1e-5f);
  __syncthreads();
#pragma unroll
  for (int i = 0; i < 4; ++i) {
    int d = t + 256 * i;
    xn[d] = (v[i] - mu) * rstd * ln_g[d] + ln_b[d];
  }
  __syncthreads();

  const float cw0 = conv_w[0], cw1 = conv_w[1], cw2 = conv_w[2], cw3 = conv_w[3];
  const float cb = conv_b[0];
#pragma unroll
  for (int i = 0; i < 4; ++i) {
    int d = t + 256 * i;
    float acc = cb;
    if (d - 3 >= 0) acc += cw0 * xn[d - 3];
    if (d - 2 >= 0) acc += cw1 * xn[d - 2];
    if (d - 1 >= 0) acc += cw2 * xn[d - 1];
    acc += cw3 * xn[d];
    float y = acc * sigmoidf_(acc);   // SiLU
    xcb[(long)row * DD + d] = f2bf(y);
  }
}

// ---------------------------------------------------------------------------
// Kernel 2: f32 -> bf16 conversion (h_prev)
// ---------------------------------------------------------------------------
__global__ __launch_bounds__(256) void cvt_bf16_kernel(
    const float* __restrict__ src, unsigned short* __restrict__ dst, long n) {
  long i = ((long)blockIdx.x * blockDim.x + threadIdx.x) * 4;
  if (i + 3 < n) {
    float4 v = *(const float4*)(src + i);
    dst[i + 0] = f2bf(v.x);
    dst[i + 1] = f2bf(v.y);
    dst[i + 2] = f2bf(v.z);
    dst[i + 3] = f2bf(v.w);
  }
}

// ---------------------------------------------------------------------------
// Kernel 3: weight pack: dst[b][n][k] = bf16(src[b][k][n]) (zero pad OOB)
// dst dims [bat][Cp][Rp], src dims [bat][R][C]
// ---------------------------------------------------------------------------
__global__ __launch_bounds__(256) void pack_transpose_kernel(
    const float* __restrict__ src, unsigned short* __restrict__ dst,
    int R, int C, int Rp, int Cp, int bat) {
  long total = (long)bat * Cp * Rp;
  long i = (long)blockIdx.x * blockDim.x + threadIdx.x;
  if (i >= total) return;
  int k = (int)(i % Rp);
  long t = i / Rp;
  int n = (int)(t % Cp);
  int b = (int)(t / Cp);
  unsigned short v = 0;
  if (n < C && k < R) v = f2bf(src[((long)b * R + k) * C + n]);
  dst[i] = v;
}

// ---------------------------------------------------------------------------
// Kernel 4: fused gate GEMMs + exponential gating + GroupNorm.
// grid = (B/16, NH). Block = 256 (8 waves). Wave w owns 16-col tile w of the
// 128-col head block. K = 128 for both x-part and h-part.
// ---------------------------------------------------------------------------
#define LDSP 132
__global__ __launch_bounds__(256) void gates_kernel(
    const unsigned short* __restrict__ xb, const unsigned short* __restrict__ xcb,
    const unsigned short* __restrict__ hb,
    const unsigned short* __restrict__ wzp, const unsigned short* __restrict__ wip,
    const unsigned short* __restrict__ wfp, const unsigned short* __restrict__ wop,
    const unsigned short* __restrict__ rzp, const unsigned short* __restrict__ rip,
    const unsigned short* __restrict__ rfp, const unsigned short* __restrict__ rop,
    const float* __restrict__ wz_b, const float* __restrict__ wi_b,
    const float* __restrict__ wf_b, const float* __restrict__ wo_b,
    const float* __restrict__ rz_b, const float* __restrict__ ri_b,
    const float* __restrict__ rf_b, const float* __restrict__ ro_b,
    const float* __restrict__ c_prev, const float* __restrict__ n_prev,
    const float* __restrict__ m_prev,
    const float* __restrict__ gn_g, const float* __restrict__ gn_b,
    float* __restrict__ h_out, float* __restrict__ c_out,
    float* __restrict__ n_out, float* __restrict__ m_out,
    unsigned short* __restrict__ hnb) {
  __shared__ float tile[16 * LDSP];
  __shared__ float psum[256], psq[256];
  __shared__ float smu[16], srstd[16];

  const int wv = threadIdx.x >> 5;
  const int lane = threadIdx.x & 31;
  const int hi = lane >> 4;
  const int ln = lane & 15;
  const int rowBase = blockIdx.x * 16;
  const int head = blockIdx.y;
  const int headOff = head * DB;
  const long wOff = (long)head * DB * DB;
  const int nLoc = wv * 16 + ln;

  v8f az = {}, ai = {}, af = {}, ao = {};
#pragma unroll
  for (int kt = 0; kt < 4; ++kt) {
    const int kb = kt * 32;
    v16bf a_x  = load_a(xb,  DD, rowBase + ln, headOff + kb, hi);
    v16bf a_xc = load_a(xcb, DD, rowBase + ln, headOff + kb, hi);
    v16bf a_h  = load_a(hb,  HH, rowBase + ln, headOff + kb, hi);
    az = WMMA_BF16(a_x,  load_b(wzp + wOff, DB, nLoc, kb, hi), az);
    ao = WMMA_BF16(a_x,  load_b(wop + wOff, DB, nLoc, kb, hi), ao);
    ai = WMMA_BF16(a_xc, load_b(wip + wOff, DB, nLoc, kb, hi), ai);
    af = WMMA_BF16(a_xc, load_b(wfp + wOff, DB, nLoc, kb, hi), af);
    az = WMMA_BF16(a_h,  load_b(rzp + wOff, DB, nLoc, kb, hi), az);
    ao = WMMA_BF16(a_h,  load_b(rop + wOff, DB, nLoc, kb, hi), ao);
    ai = WMMA_BF16(a_h,  load_b(rip + wOff, DB, nLoc, kb, hi), ai);
    af = WMMA_BF16(a_h,  load_b(rfp + wOff, DB, nLoc, kb, hi), af);
  }

  // Epilogue: exponential gating (per-lane, C layout: VGPR j -> M=j+8*hi)
  const int cGlob = headOff + nLoc;
  const float bz = wz_b[cGlob] + rz_b[cGlob];
  const float bi = wi_b[cGlob] + ri_b[cGlob];
  const float bf = wf_b[cGlob] + rf_b[cGlob];
  const float bo = wo_b[cGlob] + ro_b[cGlob];
#pragma unroll
  for (int j = 0; j < 8; ++j) {
    const int lr = j + hi * 8;
    const long idx = (long)(rowBase + lr) * HH + cGlob;
    float z  = tanhf(az[j] + bz);
    float o  = sigmoidf_(ao[j] + bo);
    float it = ai[j] + bi;
    float ft = af[j] + bf;
    float mp = m_prev[idx];
    float mt = fmaxf(ft + mp, it);
    float ig = expf(it - mt);
    float fg = expf(ft + mp - mt);
    float ct = fg * c_prev[idx] + ig * z;
    float nt = fg * n_prev[idx] + ig;
    float ht = o * ct / (nt + 1e-6f);
    h_out[idx] = ht;
    c_out[idx] = ct;
    n_out[idx] = nt;
    m_out[idx] = mt;
    tile[lr * LDSP + nLoc] = ht;
  }
  __syncthreads();

  // GroupNorm over the 128-channel head block (one group) per row.
  {
    const int r = threadIdx.x >> 4;
    const int c0 = threadIdx.x & 15;
    float s = 0.f, q = 0.f;
#pragma unroll
    for (int i = 0; i < 8; ++i) {
      float v = tile[r * LDSP + c0 + 16 * i];
      s += v; q += v * v;
    }
    psum[threadIdx.x] = s;
    psq[threadIdx.x] = q;
  }
  __syncthreads();
  if (threadIdx.x < 16) {
    float s = 0.f, q = 0.f;
#pragma unroll
    for (int i = 0; i < 16; ++i) { s += psum[threadIdx.x * 16 + i]; q += psq[threadIdx.x * 16 + i]; }
    float mu = s * (1.f / (float)DB);
    float var = q * (1.f / (float)DB) - mu * mu;
    smu[threadIdx.x] = mu;
    srstd[threadIdx.x] = rsqrtf(var + 1e-5f);
  }
  __syncthreads();
#pragma unroll
  for (int i = 0; i < 8; ++i) {
    int e = threadIdx.x + 256 * i;         // 2048 elements
    int r = e >> 7, c = e & 127;
    float v = (tile[r * LDSP + c] - smu[r]) * srstd[r];
    v = v * gn_g[headOff + c] + gn_b[headOff + c];
    hnb[(long)(rowBase + r) * HH + headOff + c] = f2bf(v);
  }
}

// ---------------------------------------------------------------------------
// Kernel 5: up-projection pair + GELU gating:  g = (hn@upl+b) * gelu(hn@upr+b)
// grid = (B/16, 11). Wave w owns 16-col tile blockIdx.y*8+w of PROJP cols.
// ---------------------------------------------------------------------------
__global__ __launch_bounds__(256) void up_kernel(
    const unsigned short* __restrict__ hnb, const unsigned short* __restrict__ uplT,
    const unsigned short* __restrict__ uprT, const float* __restrict__ upl_b,
    const float* __restrict__ upr_b, unsigned short* __restrict__ gb) {
  const int wv = threadIdx.x >> 5;
  const int lane = threadIdx.x & 31;
  const int hi = lane >> 4;
  const int ln = lane & 15;
  const int rowBase = blockIdx.x * 16;
  const int colBase = (blockIdx.y * 8 + wv) * 16;
  if (colBase >= PROJP) return;   // uniform per wave
  const int col = colBase + ln;

  v8f al = {}, ar = {};
  for (int kt = 0; kt < 32; ++kt) {
    const int kb = kt * 32;
    v16bf a = load_a(hnb, HH, rowBase + ln, kb, hi);
    al = WMMA_BF16(a, load_b(uplT, HH, col, kb, hi), al);
    ar = WMMA_BF16(a, load_b(uprT, HH, col, kb, hi), ar);
  }
#pragma unroll
  for (int j = 0; j < 8; ++j) {
    const int row = rowBase + j + hi * 8;
    float g = 0.f;
    if (col < PROJ) {
      float vl = al[j] + upl_b[col];
      float vr = ar[j] + upr_b[col];
      float gel = vr * 0.5f * (1.f + erff(vr * 0.70710678118654752f));
      g = vl * gel;
    }
    gb[(long)row * PROJP + col] = f2bf(g);
  }
}

// ---------------------------------------------------------------------------
// Kernel 6: down-projection + bias + residual:  out = g @ dn + dn_b + x
// grid = (B/16, 8). K = PROJP = 43 * 32 (zero padded).
// ---------------------------------------------------------------------------
__global__ __launch_bounds__(256) void down_kernel(
    const unsigned short* __restrict__ gb, const unsigned short* __restrict__ dnT,
    const float* __restrict__ dn_b, const float* __restrict__ x,
    float* __restrict__ out) {
  const int wv = threadIdx.x >> 5;
  const int lane = threadIdx.x & 31;
  const int hi = lane >> 4;
  const int ln = lane & 15;
  const int rowBase = blockIdx.x * 16;
  const int colBase = (blockIdx.y * 8 + wv) * 16;
  const int col = colBase + ln;

  v8f acc = {};
  for (int kt = 0; kt < 43; ++kt) {
    const int kb = kt * 32;
    v16bf a = load_a(gb, PROJP, rowBase + ln, kb, hi);
    acc = WMMA_BF16(a, load_b(dnT, PROJP, col, kb, hi), acc);
  }
  const float db = dn_b[col];
#pragma unroll
  for (int j = 0; j < 8; ++j) {
    const long idx = (long)(rowBase + j + hi * 8) * DD + col;
    out[idx] = acc[j] + db + x[idx];
  }
}

// ---------------------------------------------------------------------------
// Launcher
// ---------------------------------------------------------------------------
extern "C" void kernel_launch(void* const* d_in, const int* in_sizes, int n_in,
                              void* d_out, int out_size, void* d_ws, size_t ws_size,
                              hipStream_t stream) {
  const float* x      = (const float*)d_in[0];
  const float* h_prev = (const float*)d_in[1];
  const float* c_prev = (const float*)d_in[2];
  const float* n_prev = (const float*)d_in[3];
  const float* m_prev = (const float*)d_in[4];
  const float* ln_g   = (const float*)d_in[5];
  const float* ln_b   = (const float*)d_in[6];
  const float* conv_w = (const float*)d_in[7];
  const float* conv_b = (const float*)d_in[8];
  const float* gn_g   = (const float*)d_in[9];
  const float* gn_b   = (const float*)d_in[10];
  const float* upl_w  = (const float*)d_in[11];
  const float* upl_b  = (const float*)d_in[12];
  const float* upr_w  = (const float*)d_in[13];
  const float* upr_b  = (const float*)d_in[14];
  const float* dn_w   = (const float*)d_in[15];
  const float* dn_b   = (const float*)d_in[16];
  const float* wz_w = (const float*)d_in[17]; const float* wz_b = (const float*)d_in[18];
  const float* wi_w = (const float*)d_in[19]; const float* wi_b = (const float*)d_in[20];
  const float* wf_w = (const float*)d_in[21]; const float* wf_b = (const float*)d_in[22];
  const float* wo_w = (const float*)d_in[23]; const float* wo_b = (const float*)d_in[24];
  const float* rz_w = (const float*)d_in[25]; const float* rz_b = (const float*)d_in[26];
  const float* ri_w = (const float*)d_in[27]; const float* ri_b = (const float*)d_in[28];
  const float* rf_w = (const float*)d_in[29]; const float* rf_b = (const float*)d_in[30];
  const float* ro_w = (const float*)d_in[31]; const float* ro_b = (const float*)d_in[32];

  float* out_x = (float*)d_out;
  float* out_h = out_x + (long)BB * DD;
  float* out_c = out_h + (long)BB * HH;
  float* out_n = out_c + (long)BB * HH;
  float* out_m = out_n + (long)BB * HH;

  // Workspace carve-out (256B aligned)
  char* ws = (char*)d_ws;
  auto carve = [&](size_t bytes) -> char* {
    char* p = ws;
    ws += (bytes + 255) & ~(size_t)255;
    return p;
  };
  unsigned short* xb   = (unsigned short*)carve((size_t)BB * DD * 2);
  unsigned short* xcb  = (unsigned short*)carve((size_t)BB * DD * 2);
  unsigned short* hb   = (unsigned short*)carve((size_t)BB * HH * 2);
  unsigned short* hnb  = (unsigned short*)carve((size_t)BB * HH * 2);
  unsigned short* gb   = (unsigned short*)carve((size_t)BB * PROJP * 2);
  unsigned short* wzp  = (unsigned short*)carve((size_t)NHH * DB * DB * 2);
  unsigned short* wip  = (unsigned short*)carve((size_t)NHH * DB * DB * 2);
  unsigned short* wfp  = (unsigned short*)carve((size_t)NHH * DB * DB * 2);
  unsigned short* wop  = (unsigned short*)carve((size_t)NHH * DB * DB * 2);
  unsigned short* rzp  = (unsigned short*)carve((size_t)NHH * DB * DB * 2);
  unsigned short* rip  = (unsigned short*)carve((size_t)NHH * DB * DB * 2);
  unsigned short* rfp  = (unsigned short*)carve((size_t)NHH * DB * DB * 2);
  unsigned short* rop  = (unsigned short*)carve((size_t)NHH * DB * DB * 2);
  unsigned short* uplT = (unsigned short*)carve((size_t)PROJP * HH * 2);
  unsigned short* uprT = (unsigned short*)carve((size_t)PROJP * HH * 2);
  unsigned short* dnT  = (unsigned short*)carve((size_t)DD * PROJP * 2);

  // Stage 0: activation prep
  prep_x_kernel<<<BB, 256, 0, stream>>>(x, ln_g, ln_b, conv_w, conv_b, xb, xcb);
  cvt_bf16_kernel<<<(BB * HH) / 1024, 256, 0, stream>>>(h_prev, hb, (long)BB * HH);

  // Stage 0b: weight packing (transpose to [N][K] bf16, zero padded)
  const int gateBlocks = (NHH * DB * DB + 255) / 256;
  pack_transpose_kernel<<<gateBlocks, 256, 0, stream>>>(wz_w, wzp, DB, DB, DB, DB, NHH);
  pack_transpose_kernel<<<gateBlocks, 256, 0, stream>>>(wi_w, wip, DB, DB, DB, DB, NHH);
  pack_transpose_kernel<<<gateBlocks, 256, 0, stream>>>(wf_w, wfp, DB, DB, DB, DB, NHH);
  pack_transpose_kernel<<<gateBlocks, 256, 0, stream>>>(wo_w, wop, DB, DB, DB, DB, NHH);
  pack_transpose_kernel<<<gateBlocks, 256, 0, stream>>>(rz_w, rzp, DB, DB, DB, DB, NHH);
  pack_transpose_kernel<<<gateBlocks, 256, 0, stream>>>(ri_w, rip, DB, DB, DB, DB, NHH);
  pack_transpose_kernel<<<gateBlocks, 256, 0, stream>>>(rf_w, rfp, DB, DB, DB, DB, NHH);
  pack_transpose_kernel<<<gateBlocks, 256, 0, stream>>>(ro_w, rop, DB, DB, DB, DB, NHH);
  const int upBlocks = ((PROJP * HH) + 255) / 256;
  pack_transpose_kernel<<<upBlocks, 256, 0, stream>>>(upl_w, uplT, HH, PROJ, HH, PROJP, 1);
  pack_transpose_kernel<<<upBlocks, 256, 0, stream>>>(upr_w, uprT, HH, PROJ, HH, PROJP, 1);
  const int dnBlocks = ((DD * PROJP) + 255) / 256;
  pack_transpose_kernel<<<dnBlocks, 256, 0, stream>>>(dn_w, dnT, PROJ, DD, PROJP, DD, 1);

  // Stage 1: fused gate GEMMs + gating + GroupNorm
  gates_kernel<<<dim3(BB / 16, NHH), 256, 0, stream>>>(
      xb, xcb, hb, wzp, wip, wfp, wop, rzp, rip, rfp, rop,
      wz_b, wi_b, wf_b, wo_b, rz_b, ri_b, rf_b, ro_b,
      c_prev, n_prev, m_prev, gn_g, gn_b,
      out_h, out_c, out_n, out_m, hnb);

  // Stage 2: up-projection pair + GELU gate
  up_kernel<<<dim3(BB / 16, (PROJP + 127) / 128), 256, 0, stream>>>(
      hnb, uplT, uprT, upl_b, upr_b, gb);

  // Stage 3: down-projection + residual
  down_kernel<<<dim3(BB / 16, DD / 128), 256, 0, stream>>>(gb, dnT, dn_b, x, out_x);
}
